// GNNStep_31190052504347
// MI455X (gfx1250) — matrix operand order repaired and checked by reading, not compiled, not run on hardware
//
#include <hip/hip_runtime.h>
#include <hip/hip_bf16.h>

// ---------------------------------------------------------------------------
// GNN message-passing step for MI455X (gfx1250), wave32 + WMMA bf16.
//
//   messages = MLP2(concat(x[row], x[col], edge_attr))   [E,3H] -> [E,H]
//   agg      = segment_sum(messages, col)                (f32 atomics)
//   out      = MLP2(concat(x, agg))                      [N,2H] -> [N,H]
//
// All GEMMs use v_wmma_f32_16x16x32_bf16 (K=32/issue, f32 accumulation).
// Each wave processes TWO 16-row M-tiles so every weight B-fragment load
// (global/L2) feeds two WMMAs -> halves L2 weight traffic vs 1 tile/wave.
// ---------------------------------------------------------------------------

#define HDIM 128

typedef __attribute__((ext_vector_type(16))) __bf16 v16bf;
typedef __attribute__((ext_vector_type(8)))  __bf16 v8bf;
typedef __attribute__((ext_vector_type(4)))  __bf16 v4bf;
typedef __attribute__((ext_vector_type(8)))  float  v8f;

union AFrag { v16bf v; v8bf h[2]; };

// Load a 16x32 bf16 A-fragment (row-major source, row stride = ldk elems).
// Contiguous K order (matches B): lanes 0-15 hold M=lane, K = 0..15;
// lanes 16-31 hold M=lane-16, K = 16..31.  (kbase = 32*ks + 16*hi)
__device__ __forceinline__ v16bf load_fragA(const __bf16* __restrict__ base,
                                            int ldk, int m, int kbase) {
    const __bf16* p = base + (size_t)m * ldk + kbase;
    AFrag f;
    f.h[0] = *(const v8bf*)(p);        // K = kbase .. kbase+7
    f.h[1] = *(const v8bf*)(p + 8);    // K = kbase+8 .. kbase+15
    return f.v;
}

// Load a 32x16 bf16 B-fragment from transposed weights Wt[n][k].
// Contiguous K order (matches A): lanes 0-15 hold K=0..15 of column n, lanes 16-31 K=16..31,
// i.e. per lane one contiguous 32-byte run.
__device__ __forceinline__ v16bf load_fragB(const __bf16* __restrict__ wt,
                                            int ldk, int n, int kbase) {
    return *(const v16bf*)(wt + (size_t)n * ldk + kbase);
}

__device__ __forceinline__ v16bf wmma_bf16(v16bf a, v16bf b, v8f& acc) {
    acc = __builtin_amdgcn_wmma_f32_16x16x32_bf16(
              false, a, false, b, (short)0, acc, false, false);
    return a;
}

__device__ __forceinline__ v4bf cvt4(float4 f) {
    v4bf t;
    t[0] = (__bf16)f.x; t[1] = (__bf16)f.y; t[2] = (__bf16)f.z; t[3] = (__bf16)f.w;
    return t;
}

// ---------------------------------------------------------------------------
__global__ void zero_f32_kernel(float* __restrict__ p, int n) {
    int i = blockIdx.x * blockDim.x + threadIdx.x;
    if (i < n) p[i] = 0.0f;
}

// Wt[n*K + k] = (bf16) W[k*Nn + n]   (W is [K, Nn] row-major)
__global__ void transpose_cvt_kernel(const float* __restrict__ W,
                                     __bf16* __restrict__ Wt, int K, int Nn) {
    int i = blockIdx.x * blockDim.x + threadIdx.x;
    if (i >= K * Nn) return;
    int n = i / K;
    int k = i - n * K;
    Wt[i] = (__bf16)W[(size_t)k * Nn + n];
}

// ---------------------------------------------------------------------------
// Edge MLP + scatter-add.  One wave32 per 32-edge tile (2 M-tiles).
// Layer1: [32,384]x[384,128]  -> 8 nt x 12 ks x 2 mt WMMA (192)
// Layer2: [32,128]x[128,128]  -> 8 nt x  4 ks x 2 mt WMMA (64)
// ---------------------------------------------------------------------------
__global__ __launch_bounds__(32)
void edge_mlp_kernel(const float* __restrict__ x,
                     const int*   __restrict__ ei,     // [2*E], row then col
                     const float* __restrict__ ea,     // [E*H]
                     const __bf16* __restrict__ Wt1,   // [H][3H]
                     const float* __restrict__ be1,
                     const __bf16* __restrict__ Wt2,   // [H][H]
                     const float* __restrict__ be2,
                     float* __restrict__ agg,          // [N*H]
                     int E) {
    __shared__ __bf16 A1[32][3 * HDIM];   // gathered edge inputs (bf16), 24KB
    __shared__ __bf16 Hs[32][HDIM];       // hidden activations (bf16),  8KB
    __shared__ int    colS[32];

    const int lane = threadIdx.x;
    const int base = blockIdx.x * 32;
    const int nlo  = lane & 15;
    const int hi   = lane >> 4;

    colS[lane] = ei[E + base + lane];

    // Gather + f32->bf16 convert: each lane handles 4 contiguous elems / row.
    for (int e = 0; e < 32; ++e) {
        const int r = ei[base + e];
        const int c = ei[E + base + e];
        float4 xr = *(const float4*)(x + (size_t)r * HDIM + lane * 4);
        float4 xc = *(const float4*)(x + (size_t)c * HDIM + lane * 4);
        float4 av = *(const float4*)(ea + (size_t)(base + e) * HDIM + lane * 4);
        *(v4bf*)&A1[e][lane * 4]            = cvt4(xr);
        *(v4bf*)&A1[e][HDIM + lane * 4]     = cvt4(xc);
        *(v4bf*)&A1[e][2 * HDIM + lane * 4] = cvt4(av);
    }
    __syncthreads();

    // ---- Layer 1: h = relu(A1 @ We1 + be1) -------------------------------
    #pragma unroll
    for (int nt = 0; nt < 8; ++nt) {
        v8f acc0 = {}, acc1 = {};
        #pragma unroll
        for (int ks = 0; ks < 12; ++ks) {
            v16bf b  = load_fragB(Wt1, 3 * HDIM, nt * 16 + nlo, ks * 32 + hi * 16);
            v16bf a0 = load_fragA(&A1[0][0],  3 * HDIM, nlo, ks * 32 + hi * 16);
            v16bf a1 = load_fragA(&A1[16][0], 3 * HDIM, nlo, ks * 32 + hi * 16);
            wmma_bf16(a0, b, acc0);
            wmma_bf16(a1, b, acc1);
        }
        const float bias = be1[nt * 16 + nlo];
        #pragma unroll
        for (int i = 0; i < 8; ++i) {          // C layout: M=i+8*hi, N=nlo
            float v0 = acc0[i] + bias;
            float v1 = acc1[i] + bias;
            Hs[i + hi * 8][nt * 16 + nlo]      = (__bf16)(v0 > 0.0f ? v0 : 0.0f);
            Hs[16 + i + hi * 8][nt * 16 + nlo] = (__bf16)(v1 > 0.0f ? v1 : 0.0f);
        }
    }
    __syncthreads();

    // ---- Layer 2 + scatter: agg[col] += h @ We2 + be2 --------------------
    #pragma unroll
    for (int nt = 0; nt < 8; ++nt) {
        v8f acc0 = {}, acc1 = {};
        #pragma unroll
        for (int ks = 0; ks < 4; ++ks) {
            v16bf b  = load_fragB(Wt2, HDIM, nt * 16 + nlo, ks * 32 + hi * 16);
            v16bf a0 = load_fragA(&Hs[0][0],  HDIM, nlo, ks * 32 + hi * 16);
            v16bf a1 = load_fragA(&Hs[16][0], HDIM, nlo, ks * 32 + hi * 16);
            wmma_bf16(a0, b, acc0);
            wmma_bf16(a1, b, acc1);
        }
        const float bias = be2[nt * 16 + nlo];
        #pragma unroll
        for (int i = 0; i < 8; ++i) {
            const int m = i + hi * 8;
            atomicAdd(&agg[(size_t)colS[m]      * HDIM + nt * 16 + nlo], acc0[i] + bias);
            atomicAdd(&agg[(size_t)colS[m + 16] * HDIM + nt * 16 + nlo], acc1[i] + bias);
        }
    }
}

// ---------------------------------------------------------------------------
// Node MLP.  One wave32 per 32-node tile (2 M-tiles).
// Layer1: [32,256]x[256,128] -> 8 nt x 8 ks x 2 mt WMMA (128)
// Layer2: [32,128]x[128,128] -> 8 nt x 4 ks x 2 mt WMMA (64)
// ---------------------------------------------------------------------------
__global__ __launch_bounds__(32)
void node_mlp_kernel(const float* __restrict__ x,
                     const float* __restrict__ agg,
                     const __bf16* __restrict__ Wt1,   // [H][2H]
                     const float* __restrict__ bn1,
                     const __bf16* __restrict__ Wt2,   // [H][H]
                     const float* __restrict__ bn2,
                     float* __restrict__ out,
                     int N) {
    __shared__ __bf16 A2[32][2 * HDIM];   // 16KB
    __shared__ __bf16 Hs[32][HDIM];       //  8KB

    const int lane  = threadIdx.x;
    const int nbase = blockIdx.x * 32;
    const int nlo   = lane & 15;
    const int hi    = lane >> 4;

    for (int e = 0; e < 32; ++e) {
        int node = nbase + e;
        if (node >= N) node = N - 1;                   // clamp (dup rows, stores guarded)
        float4 xv = *(const float4*)(x   + (size_t)node * HDIM + lane * 4);
        float4 gv = *(const float4*)(agg + (size_t)node * HDIM + lane * 4);
        *(v4bf*)&A2[e][lane * 4]        = cvt4(xv);
        *(v4bf*)&A2[e][HDIM + lane * 4] = cvt4(gv);
    }
    __syncthreads();

    #pragma unroll
    for (int nt = 0; nt < 8; ++nt) {
        v8f acc0 = {}, acc1 = {};
        #pragma unroll
        for (int ks = 0; ks < 8; ++ks) {
            v16bf b  = load_fragB(Wt1, 2 * HDIM, nt * 16 + nlo, ks * 32 + hi * 16);
            v16bf a0 = load_fragA(&A2[0][0],  2 * HDIM, nlo, ks * 32 + hi * 16);
            v16bf a1 = load_fragA(&A2[16][0], 2 * HDIM, nlo, ks * 32 + hi * 16);
            wmma_bf16(a0, b, acc0);
            wmma_bf16(a1, b, acc1);
        }
        const float bias = bn1[nt * 16 + nlo];
        #pragma unroll
        for (int i = 0; i < 8; ++i) {
            float v0 = acc0[i] + bias;
            float v1 = acc1[i] + bias;
            Hs[i + hi * 8][nt * 16 + nlo]      = (__bf16)(v0 > 0.0f ? v0 : 0.0f);
            Hs[16 + i + hi * 8][nt * 16 + nlo] = (__bf16)(v1 > 0.0f ? v1 : 0.0f);
        }
    }
    __syncthreads();

    #pragma unroll
    for (int nt = 0; nt < 8; ++nt) {
        v8f acc0 = {}, acc1 = {};
        #pragma unroll
        for (int ks = 0; ks < 4; ++ks) {
            v16bf b  = load_fragB(Wt2, HDIM, nt * 16 + nlo, ks * 32 + hi * 16);
            v16bf a0 = load_fragA(&Hs[0][0],  HDIM, nlo, ks * 32 + hi * 16);
            v16bf a1 = load_fragA(&Hs[16][0], HDIM, nlo, ks * 32 + hi * 16);
            wmma_bf16(a0, b, acc0);
            wmma_bf16(a1, b, acc1);
        }
        const float bias = bn2[nt * 16 + nlo];
        #pragma unroll
        for (int i = 0; i < 8; ++i) {
            const int n0 = nbase + i + hi * 8;
            const int n1 = n0 + 16;
            if (n0 < N) out[(size_t)n0 * HDIM + nt * 16 + nlo] = acc0[i] + bias;
            if (n1 < N) out[(size_t)n1 * HDIM + nt * 16 + nlo] = acc1[i] + bias;
        }
    }
}

// ---------------------------------------------------------------------------
extern "C" void kernel_launch(void* const* d_in, const int* in_sizes, int n_in,
                              void* d_out, int out_size, void* d_ws, size_t ws_size,
                              hipStream_t stream) {
    const float* x   = (const float*)d_in[0];
    const int*   ei  = (const int*)  d_in[1];
    const float* ea  = (const float*)d_in[2];
    const float* We1 = (const float*)d_in[3];
    const float* be1 = (const float*)d_in[4];
    const float* We2 = (const float*)d_in[5];
    const float* be2 = (const float*)d_in[6];
    const float* Wn1 = (const float*)d_in[7];
    const float* bn1 = (const float*)d_in[8];
    const float* Wn2 = (const float*)d_in[9];
    const float* bn2 = (const float*)d_in[10];
    float* out = (float*)d_out;

    const int N = in_sizes[0] / HDIM;
    const int E = in_sizes[1] / 2;

    // Workspace carve-up (256B aligned): agg + 4 transposed bf16 weights.
    char* ws = (char*)d_ws;
    size_t off = 0;
    auto carve = [&](size_t bytes) {
        void* p = ws + off;
        off = (off + bytes + 255) & ~(size_t)255;
        return p;
    };
    float*  agg  = (float*) carve((size_t)N * HDIM * sizeof(float));
    __bf16* Wt1  = (__bf16*)carve((size_t)3 * HDIM * HDIM * sizeof(__bf16)); // [H][3H]
    __bf16* Wt2  = (__bf16*)carve((size_t)HDIM * HDIM * sizeof(__bf16));     // [H][H]
    __bf16* Wnt1 = (__bf16*)carve((size_t)2 * HDIM * HDIM * sizeof(__bf16)); // [H][2H]
    __bf16* Wnt2 = (__bf16*)carve((size_t)HDIM * HDIM * sizeof(__bf16));     // [H][H]
    (void)ws_size; (void)n_in; (void)out_size;

    // 1) zero the scatter accumulator
    {
        int n = N * HDIM;
        zero_f32_kernel<<<(n + 255) / 256, 256, 0, stream>>>(agg, n);
    }
    // 2) transpose + convert weights to bf16
    {
        int n1 = 3 * HDIM * HDIM;
        transpose_cvt_kernel<<<(n1 + 255) / 256, 256, 0, stream>>>(We1, Wt1, 3 * HDIM, HDIM);
        int n2 = HDIM * HDIM;
        transpose_cvt_kernel<<<(n2 + 255) / 256, 256, 0, stream>>>(We2, Wt2, HDIM, HDIM);
        int n3 = 2 * HDIM * HDIM;
        transpose_cvt_kernel<<<(n3 + 255) / 256, 256, 0, stream>>>(Wn1, Wnt1, 2 * HDIM, HDIM);
        transpose_cvt_kernel<<<(n2 + 255) / 256, 256, 0, stream>>>(Wn2, Wnt2, HDIM, HDIM);
    }
    // 3) edge MLP + scatter-add (E assumed multiple of 32; 800000/32 = 25000)
    edge_mlp_kernel<<<E / 32, 32, 0, stream>>>(x, ei, ea, Wt1, be1, Wt2, be2, agg, E);
    // 4) node MLP
    node_mlp_kernel<<<(N + 31) / 32, 32, 0, stream>>>(x, agg, Wnt1, bn1, Wnt2, bn2, out, N);
}
